// HGT_12859132084346
// MI455X (gfx1250) — matrix operand order repaired and checked by reading, not compiled, not run on hardware
//
#include <hip/hip_runtime.h>
#include <hip/hip_bf16.h>
#include <math.h>

// ---------------------------------------------------------------------------
// HGT forward for MI455X (gfx1250, wave32, WMMA).
// Per-edge rel transforms are folded into node-level GEMMs:
//   k_rel_node = x @ (k_W @ blockdiag(a_rel)) + (k_b @ blockdiag(a_rel))
// so all matrix math runs through v_wmma_f32_16x16x32_f16 (f32 accumulate),
// and per-edge kernels are pure memory ops (gather + f32 atomics in L2).
// GEMM: 8 waves/block, each wave owns TWO 16x16 C tiles sharing one A
// fragment (2 back-to-back independent WMMAs per K-step); K-tail hoisted
// out of the loop so the steady state is branch-free.
// ---------------------------------------------------------------------------

#define NC_   100000
#define NQ_   20000
#define FC_   300
#define FQ_   768
#define HID_  128
#define NH_   4
#define HD_   32
#define NOUT_ 8
#define NE_   400000
#define NL_   2
#define NNT_  2
#define NET_  3

typedef __attribute__((ext_vector_type(16))) _Float16 v16h;
typedef __attribute__((ext_vector_type(8)))  float    v8f;
typedef __attribute__((ext_vector_type(4)))  float    v4f;

// ---------------------------------------------------------------------------
// Pre-swizzle a f32 weight matrix W[K,128] into WMMA B-fragment order (f16).
// Fragment layout (16-bit B, 32x16): lane = n + 16*s holds column n,
// element i<8 -> k = kt*32 + 8*s + i, i>=8 -> k = kt*32 + 16 + 8*s + (i-8).
// ---------------------------------------------------------------------------
__global__ void hgt_pack_b(const float* __restrict__ W, int K,
                           _Float16* __restrict__ Bp) {
  int idx = blockIdx.x * blockDim.x + threadIdx.x;
  int ktiles = (K + 31) >> 5;
  int total = ktiles * 8 * 32;
  if (idx >= total) return;
  int lane = idx & 31;
  int frag = idx >> 5;
  int ntile = frag & 7;
  int kt = frag >> 3;
  int s = lane >> 4;
  int n = ntile * 16 + (lane & 15);
  _Float16* out = Bp + ((size_t)idx << 4);
#pragma unroll
  for (int i = 0; i < 16; ++i) {
    int k = kt * 32 + ((i < 8) ? (8 * s + i) : (16 + 8 * s + (i - 8)));
    out[i] = (_Float16)((k < K) ? W[(size_t)k * HID_ + n] : 0.0f);
  }
}

// ---------------------------------------------------------------------------
// WMMA GEMM: C[M,128] = A[M,K] @ B + bias, N fixed to 128, M % 32 == 0.
// Block = 256 threads = 8 waves covering a 32x128 slab:
//   wave w: mtile = blockIdx.x*2 + (w>>2), ntiles 2*(w&3) and 2*(w&3)+1.
// A fragment (16-bit 16x32): lane m+16s holds row m, K runs [8s,8s+8) and
// [16+8s,16+8s+8) -> two contiguous float4-pair loads per K-step, shared by
// both WMMAs. modes: 0 = +bias, 1 = relu(+bias),
// 2 = skip-mix: C = g*(acc+bias)+(1-g)*Xold (in-place safe, one thread/elem).
// ---------------------------------------------------------------------------
__global__ __launch_bounds__(256) void hgt_wmma_gemm_n128(
    const float* __restrict__ A, int K, int lda,
    const _Float16* __restrict__ Bp,
    const float* __restrict__ bias,
    float* __restrict__ Cout,
    const float* __restrict__ Xold,
    const float* __restrict__ gate,
    int mode) {
  const int wavei = threadIdx.x >> 5;                 // 0..7
  const int lane  = threadIdx.x & 31;
  const int mtile = blockIdx.x * 2 + (wavei >> 2);
  const int nt0   = (wavei & 3) * 2;                  // first of two ntiles
  const int s     = lane >> 4;
  const int rowA  = mtile * 16 + (lane & 15);
  const float* arow = A + (size_t)rowA * lda;
  const int kfull = K >> 5;

  v8f acc0 = {0.f, 0.f, 0.f, 0.f, 0.f, 0.f, 0.f, 0.f};
  v8f acc1 = {0.f, 0.f, 0.f, 0.f, 0.f, 0.f, 0.f, 0.f};

  for (int kt = 0; kt < kfull; ++kt) {
    const int k0 = (kt << 5) + 8 * s;                 // run 1: [k0, k0+8)
    __builtin_prefetch(arow + k0 + 32, 0, 3);         // next K-step (spec. ok)
    const v4f p0 = *(const v4f*)(arow + k0);
    const v4f p1 = *(const v4f*)(arow + k0 + 4);
    const v4f p2 = *(const v4f*)(arow + k0 + 16);     // run 2: [k0+16, k0+24)
    const v4f p3 = *(const v4f*)(arow + k0 + 20);
    v16h a;
#pragma unroll
    for (int t = 0; t < 4; ++t) {
      a[t]      = (_Float16)p0[t];
      a[4 + t]  = (_Float16)p1[t];
      a[8 + t]  = (_Float16)p2[t];
      a[12 + t] = (_Float16)p3[t];
    }
    const _Float16* bb = Bp + (((size_t)(kt * 8 + nt0) * 32 + lane) << 4);
    const v16h b0 = *(const v16h*)bb;
    const v16h b1 = *(const v16h*)(bb + 512);         // next ntile fragment
    acc0 = __builtin_amdgcn_wmma_f32_16x16x32_f16(false, a, false, b0,
                                                  (short)0, acc0, false, false);
    acc1 = __builtin_amdgcn_wmma_f32_16x16x32_f16(false, a, false, b1,
                                                  (short)0, acc1, false, false);
  }

  if (K & 31) {                                       // single K tail (K=300)
    const int k0 = (kfull << 5) + 8 * s;
    const int k1 = k0 + 16;
    v16h a;
#pragma unroll
    for (int t = 0; t < 8; ++t) {
      a[t]     = (_Float16)((k0 + t < K) ? arow[k0 + t] : 0.0f);
      a[8 + t] = (_Float16)((k1 + t < K) ? arow[k1 + t] : 0.0f);
    }
    const _Float16* bb = Bp + (((size_t)(kfull * 8 + nt0) * 32 + lane) << 4);
    const v16h b0 = *(const v16h*)bb;
    const v16h b1 = *(const v16h*)(bb + 512);
    acc0 = __builtin_amdgcn_wmma_f32_16x16x32_f16(false, a, false, b0,
                                                  (short)0, acc0, false, false);
    acc1 = __builtin_amdgcn_wmma_f32_16x16x32_f16(false, a, false, b1,
                                                  (short)0, acc1, false, false);
  }

  // C layout: VGPR r -> row mtile*16 + r + 8*s, col ntile*16 + (lane&15)
  const int mrow0 = mtile * 16 + 8 * s;
  const float gv = (mode == 2) ? 1.0f / (1.0f + expf(-gate[0])) : 0.0f;
#pragma unroll
  for (int t2 = 0; t2 < 2; ++t2) {
    const v8f acc = t2 ? acc1 : acc0;
    const int ncol = (nt0 + t2) * 16 + (lane & 15);
    const float bv = bias ? bias[ncol] : 0.0f;
    if (mode == 2) {
#pragma unroll
      for (int r = 0; r < 8; ++r) {
        const size_t idx = (size_t)(mrow0 + r) * HID_ + ncol;
        const float o = acc[r] + bv;
        Cout[idx] = gv * o + (1.0f - gv) * Xold[idx];
      }
    } else {
#pragma unroll
      for (int r = 0; r < 8; ++r) {
        float o = acc[r] + bv;
        if (mode == 1) o = fmaxf(o, 0.0f);
        Cout[(size_t)(mrow0 + r) * HID_ + ncol] = o;
      }
    }
  }
}

// ---------------------------------------------------------------------------
// Fold per-head rel matrices into the projection weight:
//   Wout[i, h*32+f] = sum_d W[i, h*32+d] * rel[h,d,f]   (row 128 -> bias)
// ---------------------------------------------------------------------------
__global__ void hgt_combine_rel(const float* __restrict__ W,
                                const float* __restrict__ bvec,
                                const float* __restrict__ rel,   // [H,32,32]
                                float* __restrict__ Wout,
                                float* __restrict__ bout) {
  int j = blockIdx.x * blockDim.x + threadIdx.x;   // 129*128 threads
  if (j >= 129 * HID_) return;
  int col = j & (HID_ - 1);
  int row = j >> 7;
  int h = col >> 5, f = col & 31;
  const float* relh = rel + h * (HD_ * HD_) + f;            // rel[h,d,f]
  const float* srcp = (row < HID_) ? (W + (size_t)row * HID_ + h * HD_)
                                   : (bvec + h * HD_);
  float accv = 0.0f;
#pragma unroll
  for (int d = 0; d < HD_; ++d) accv += srcp[d] * relh[d * HD_];
  if (row < HID_) Wout[(size_t)row * HID_ + col] = accv;
  else            bout[col] = accv;
}

__global__ void hgt_fill(float* __restrict__ p, float v, int n) {
  int t = blockIdx.x * blockDim.x + threadIdx.x;
  if (t < n) p[t] = v;
}

__device__ __forceinline__ void hgt_atomic_max_f32(float* addr, float val) {
  // bit-trick float max: monotone in both int (>=0) and uint (<0) orderings
  if (val >= 0.0f) atomicMax((int*)addr, __float_as_int(val));
  else             atomicMin((unsigned int*)addr, __float_as_uint(val));
}

// One 32-lane wave per edge: lane j covers dims [4j,4j+4), head = j>>3.
__global__ __launch_bounds__(256) void hgt_edge_logits(
    const int* __restrict__ src, const int* __restrict__ dst,
    const float* __restrict__ Q, const float* __restrict__ Krel,
    const float* __restrict__ prel, float* __restrict__ logits,
    float* __restrict__ segmax) {
  int e = blockIdx.x * 8 + (threadIdx.x >> 5);
  if (e >= NE_) return;
  int lane = threadIdx.x & 31;
  int sI = src[e], dI = dst[e];
  float4 qv = ((const float4*)(Q    + (size_t)dI * HID_))[lane];
  float4 kv = ((const float4*)(Krel + (size_t)sI * HID_))[lane];
  float p = qv.x * kv.x + qv.y * kv.y + qv.z * kv.z + qv.w * kv.w;
  p += __shfl_down(p, 4, 8);
  p += __shfl_down(p, 2, 8);
  p += __shfl_down(p, 1, 8);
  if ((lane & 7) == 0) {
    int h = lane >> 3;
    float lg = p * prel[h] * 0.17677669529663687f;  // 1/sqrt(32)
    logits[(size_t)e * NH_ + h] = lg;
    hgt_atomic_max_f32(&segmax[(size_t)dI * NH_ + h], lg);
  }
}

__global__ void hgt_edge_exp(const int* __restrict__ dst,
                             const float* __restrict__ logits,
                             const float* __restrict__ segmax,
                             float* __restrict__ evals,
                             float* __restrict__ segsum) {
  int t = blockIdx.x * blockDim.x + threadIdx.x;
  if (t >= NE_ * NH_) return;
  int e = t >> 2, h = t & 3;
  int dI = dst[e];
  float ev = expf(logits[t] - segmax[(size_t)dI * NH_ + h]);
  evals[t] = ev;
  atomicAdd(&segsum[(size_t)dI * NH_ + h], ev);
}

__global__ __launch_bounds__(256) void hgt_edge_scatter(
    const int* __restrict__ src, const int* __restrict__ dst,
    const float* __restrict__ Vrel, const float* __restrict__ evals,
    const float* __restrict__ segsum, float* __restrict__ agg) {
  int e = blockIdx.x * 8 + (threadIdx.x >> 5);
  if (e >= NE_) return;
  int lane = threadIdx.x & 31;
  int h = lane >> 3;
  int sI = src[e], dI = dst[e];
  float alpha = evals[(size_t)e * NH_ + h] /
                (segsum[(size_t)dI * NH_ + h] + 1e-16f);
  float4 vv = ((const float4*)(Vrel + (size_t)sI * HID_))[lane];
  float* out = agg + (size_t)dI * HID_ + lane * 4;
  atomicAdd(out + 0, vv.x * alpha);
  atomicAdd(out + 1, vv.y * alpha);
  atomicAdd(out + 2, vv.z * alpha);
  atomicAdd(out + 3, vv.w * alpha);
}

__global__ void hgt_gelu(float* __restrict__ a, int n) {
  int t = blockIdx.x * blockDim.x + threadIdx.x;
  if (t >= n) return;
  float x = a[t];
  a[t] = 0.5f * x * (1.0f + erff(x * 0.70710678118654752f));
}

__global__ void hgt_out_proj(const float* __restrict__ x,
                             const float* __restrict__ W,   // [128,8]
                             const float* __restrict__ b,
                             float* __restrict__ out) {
  int t = blockIdx.x * blockDim.x + threadIdx.x;
  if (t >= NC_ * NOUT_) return;
  int i = t >> 3, o = t & 7;
  const float* xr = x + (size_t)i * HID_;
  float accv = b[o];
#pragma unroll 8
  for (int d = 0; d < HID_; ++d) accv += xr[d] * W[d * NOUT_ + o];
  out[t] = accv;
}

// ---------------------------------------------------------------------------

extern "C" void kernel_launch(void* const* d_in, const int* in_sizes, int n_in,
                              void* d_out, int out_size, void* d_ws, size_t ws_size,
                              hipStream_t stream) {
  (void)in_sizes; (void)n_in; (void)out_size; (void)ws_size;

  const float* x_context  = (const float*)d_in[0];
  const float* x_question = (const float*)d_in[1];
  const int* esrc[3] = { (const int*)d_in[2], (const int*)d_in[4], (const int*)d_in[6] };
  const int* edst[3] = { (const int*)d_in[3], (const int*)d_in[5], (const int*)d_in[7] };
  const float* inW[2] = { (const float*)d_in[8],  (const float*)d_in[10] };
  const float* inB[2] = { (const float*)d_in[9],  (const float*)d_in[11] };
  const float* kW = (const float*)d_in[12];
  const float* kB = (const float*)d_in[13];
  const float* qW = (const float*)d_in[14];
  const float* qB = (const float*)d_in[15];
  const float* vW = (const float*)d_in[16];
  const float* vB = (const float*)d_in[17];
  const float* aW = (const float*)d_in[18];
  const float* aB = (const float*)d_in[19];
  const float* skip  = (const float*)d_in[20];
  const float* a_rel = (const float*)d_in[21];
  const float* m_rel = (const float*)d_in[22];
  const float* p_rel = (const float*)d_in[23];
  const float* outW  = (const float*)d_in[24];
  const float* outB  = (const float*)d_in[25];

  // ---- workspace carve (f32, 256B aligned regions) ----
  float* ws = (float*)d_ws;
  size_t off = 0;
  auto carve = [&](size_t n) {
    float* p = ws + off;
    off += (n + 63) & ~(size_t)63;
    return p;
  };
  float* x[2];   x[0]   = carve((size_t)NC_ * HID_); x[1]   = carve((size_t)NQ_ * HID_);
  float* q[2];   q[0]   = carve((size_t)NC_ * HID_); q[1]   = carve((size_t)NQ_ * HID_);
  float* krel   = carve((size_t)NC_ * HID_);
  float* vrel   = carve((size_t)NC_ * HID_);
  float* agg[2]; agg[0] = carve((size_t)NC_ * HID_); agg[1] = carve((size_t)NQ_ * HID_);
  float* segmax = carve((size_t)NC_ * NH_);
  float* segsum = carve((size_t)NC_ * NH_);
  float* evals  = carve((size_t)NE_ * NH_);
  float* Wtmp   = carve((size_t)HID_ * HID_);
  float* btmp   = carve(HID_);
  _Float16* Bp  = (_Float16*)carve((size_t)24 * 8 * 256);  // max 24 ktiles (K=768)

  const int Ns[2] = { NC_, NQ_ };
  const int snt_tab[3] = { 1, 0, 0 };
  const int dnt_tab[3] = { 0, 0, 1 };

  auto pack = [&](const float* W, int K) {
    int total = ((K + 31) >> 5) * 8 * 32;
    hgt_pack_b<<<(total + 255) / 256, 256, 0, stream>>>(W, K, Bp);
  };
  auto gemm = [&](const float* A, int M, int K, int lda, const float* bias,
                  float* C, const float* Xold, const float* gate, int mode) {
    hgt_wmma_gemm_n128<<<M / 32, 256, 0, stream>>>(A, K, lda, Bp, bias, C,
                                                   Xold, gate, mode);
  };
  auto fill = [&](float* p, float v, int n) {
    hgt_fill<<<(n + 255) / 256, 256, 0, stream>>>(p, v, n);
  };

  // ---- input projections + relu ----
  pack(inW[0], FC_);
  gemm(x_context,  NC_, FC_, FC_, inB[0], x[0], nullptr, nullptr, 1);
  pack(inW[1], FQ_);
  gemm(x_question, NQ_, FQ_, FQ_, inB[1], x[1], nullptr, nullptr, 1);

  // ---- layers ----
  for (int l = 0; l < NL_; ++l) {
    for (int nt = 0; nt < NNT_; ++nt) {
      pack(qW + (size_t)(l * NNT_ + nt) * HID_ * HID_, HID_);
      gemm(x[nt], Ns[nt], HID_, HID_, qB + (l * NNT_ + nt) * HID_,
           q[nt], nullptr, nullptr, 0);
      fill(agg[nt], 0.0f, Ns[nt] * HID_);
    }
    for (int et = 0; et < NET_; ++et) {
      const int snt = snt_tab[et], dnt = dnt_tab[et];
      const float* arel = a_rel + (size_t)(l * NET_ + et) * NH_ * HD_ * HD_;
      const float* mrel = m_rel + (size_t)(l * NET_ + et) * NH_ * HD_ * HD_;

      // k_rel_node = x[snt] @ (k_W @ A_bd) + (k_b @ A_bd)
      hgt_combine_rel<<<129, 128, 0, stream>>>(
          kW + (size_t)(l * NNT_ + snt) * HID_ * HID_,
          kB + (l * NNT_ + snt) * HID_, arel, Wtmp, btmp);
      pack(Wtmp, HID_);
      gemm(x[snt], Ns[snt], HID_, HID_, btmp, krel, nullptr, nullptr, 0);

      // v_rel_node = x[snt] @ (v_W @ M_bd) + (v_b @ M_bd)
      hgt_combine_rel<<<129, 128, 0, stream>>>(
          vW + (size_t)(l * NNT_ + snt) * HID_ * HID_,
          vB + (l * NNT_ + snt) * HID_, mrel, Wtmp, btmp);
      pack(Wtmp, HID_);
      gemm(x[snt], Ns[snt], HID_, HID_, btmp, vrel, nullptr, nullptr, 0);

      // segment softmax + aggregation
      fill(segmax, -INFINITY, Ns[dnt] * NH_);
      fill(segsum, 0.0f,      Ns[dnt] * NH_);
      hgt_edge_logits<<<(NE_ + 7) / 8, 256, 0, stream>>>(
          esrc[et], edst[et], q[dnt], krel,
          p_rel + (l * NET_ + et) * NH_, evals, segmax);
      hgt_edge_exp<<<(NE_ * NH_ + 255) / 256, 256, 0, stream>>>(
          edst[et], evals, segmax, evals, segsum);
      hgt_edge_scatter<<<(NE_ + 7) / 8, 256, 0, stream>>>(
          esrc[et], edst[et], vrel, evals, segsum, agg[dnt]);
    }
    for (int nt = 0; nt < NNT_; ++nt) {
      hgt_gelu<<<(Ns[nt] * HID_ + 255) / 256, 256, 0, stream>>>(
          agg[nt], Ns[nt] * HID_);
      pack(aW + (size_t)(l * NNT_ + nt) * HID_ * HID_, HID_);
      // in-place skip mix: x = g*(gelu(agg)@a_W + a_b) + (1-g)*x
      gemm(agg[nt], Ns[nt], HID_, HID_, aB + (l * NNT_ + nt) * HID_,
           x[nt], x[nt], skip + l * NNT_ + nt, 2);
    }
  }

  // ---- output projection ----
  hgt_out_proj<<<(NC_ * NOUT_ + 255) / 256, 256, 0, stream>>>(
      x[0], outW, outB, (float*)d_out);
}
